// MultiHeadAttention_74277164417658
// MI455X (gfx1250) — matrix operand order repaired and checked by reading, not compiled, not run on hardware
//
#include <hip/hip_runtime.h>
#include <hip/hip_bf16.h>
#include <math.h>

#define SEQ 2048
#define DM  1024
#define NH  16
#define DEP 64

typedef __attribute__((ext_vector_type(16))) __bf16 v16bf;
typedef __attribute__((ext_vector_type(8)))  __bf16 v8bf;
typedef __attribute__((ext_vector_type(4)))  __bf16 v4bf;
typedef __attribute__((ext_vector_type(8)))  float  v8f;
typedef __attribute__((ext_vector_type(4)))  float  v4f;

// ---- WMMA helpers (CDNA5 16x16x32 bf16, f32 accumulate) -------------------
static __device__ __forceinline__ v8f wmma_bf16(v16bf a, v16bf b, v8f c) {
  return __builtin_amdgcn_wmma_f32_16x16x32_bf16(false, a, false, b, (short)0, c,
                                                 false, false);
}
// A-fragment pattern: lane holds K = {base..base+7} and {base+16..base+23}
static __device__ __forceinline__ v16bf frag_split(const __bf16* p) {
  v8bf lo = *(const v8bf*)(p);
  v8bf hi = *(const v8bf*)(p + 16);
  return __builtin_shufflevector(lo, hi, 0,1,2,3,4,5,6,7,8,9,10,11,12,13,14,15);
}
// unused helper: lane holds 16 contiguous K values
static __device__ __forceinline__ v16bf frag_contig(const __bf16* p) {
  v8bf lo = *(const v8bf*)(p);
  v8bf hi = *(const v8bf*)(p + 8);
  return __builtin_shufflevector(lo, hi, 0,1,2,3,4,5,6,7,8,9,10,11,12,13,14,15);
}

// ---- prep: f32 -> bf16 elementwise ---------------------------------------
__global__ __launch_bounds__(256) void k_cvt_bf16(const float* __restrict__ x,
                                                  __bf16* __restrict__ y) {
  size_t i = ((size_t)blockIdx.x * 256 + threadIdx.x) * 4;
  v4f v = *(const v4f*)(x + i);
  v4bf o = { (__bf16)v[0], (__bf16)v[1], (__bf16)v[2], (__bf16)v[3] };
  *(v4bf*)(y + i) = o;
}

// ---- prep: W [k][n] f32 -> W^T [n][k] bf16 -------------------------------
__global__ __launch_bounds__(256) void k_transpose_w(const float* __restrict__ w,
                                                     __bf16* __restrict__ wt) {
  __shared__ float t[16][17];
  int k0 = blockIdx.y * 16, n0 = blockIdx.x * 16;
  t[threadIdx.y][threadIdx.x] = w[(size_t)(k0 + threadIdx.y) * DM + n0 + threadIdx.x];
  __syncthreads();
  wt[(size_t)(n0 + threadIdx.y) * DM + k0 + threadIdx.x] =
      (__bf16)t[threadIdx.x][threadIdx.y];
}

// ---- GEMM: X[4096 x 1024](bf16) @ W^T + bias, 32x64 tile per wave --------
// MODE 0: Qh bf16 [B,H,S,64], scaled by 0.125 (1/sqrt(64) folded into Q)
// MODE 1: Kh bf16 [B,H,S,64]
// MODE 2: VhT bf16 [B,H,64,S]
// MODE 3: f32 out [B,S,D]
template <int MODE>
__global__ __launch_bounds__(256) void k_gemm_proj(const __bf16* __restrict__ A,
                                                   const __bf16* __restrict__ WT,
                                                   const float* __restrict__ bias,
                                                   __bf16* __restrict__ outB,
                                                   float* __restrict__ outF) {
  int wave = threadIdx.x >> 5, lane = threadIdx.x & 31;
  int g = lane >> 4, ln = lane & 15;
  int tile = blockIdx.x * 8 + wave;   // 2048 tiles: 128 row-tiles x 16 col-tiles
  int rt = tile >> 4, ct = tile & 15;

  const __bf16* Arow0 = A + (size_t)(rt * 32 + ln) * DM;        // rows r..r+15
  const __bf16* Arow1 = Arow0 + (size_t)16 * DM;                // rows r+16..r+31
  const __bf16* Brow[4];
#pragma unroll
  for (int cc = 0; cc < 4; ++cc)
    Brow[cc] = WT + (size_t)(ct * 64 + cc * 16 + ln) * DM;

  v8f acc[2][4];
#pragma unroll
  for (int rr = 0; rr < 2; ++rr)
#pragma unroll
    for (int cc = 0; cc < 4; ++cc) acc[rr][cc] = {};

#pragma unroll 2
  for (int kk = 0; kk < DM; kk += 32) {
    v16bf a0 = frag_split(Arow0 + kk + 8 * g);
    v16bf a1 = frag_split(Arow1 + kk + 8 * g);
#pragma unroll
    for (int cc = 0; cc < 4; ++cc) {
      v16bf bf = frag_split(Brow[cc] + kk + 8 * g);
      acc[0][cc] = wmma_bf16(a0, bf, acc[0][cc]);
      acc[1][cc] = wmma_bf16(a1, bf, acc[1][cc]);
    }
  }

#pragma unroll
  for (int cc = 0; cc < 4; ++cc) {
    int c = ct * 64 + cc * 16 + ln;
    float bv = bias[c];
#pragma unroll
    for (int rr = 0; rr < 2; ++rr) {
#pragma unroll
      for (int i = 0; i < 8; ++i) {
        float v = acc[rr][cc][i] + bv;
        int r = rt * 32 + rr * 16 + i + 8 * g;   // flattened row over B*S = 4096
        if (MODE == 3) {
          outF[(size_t)r * DM + c] = v;
        } else {
          int b = r >> 11, s = r & 2047, h = c >> 6, d = c & 63;
          if (MODE == 0)
            outB[((size_t)(b * NH + h) * SEQ + s) * DEP + d] = (__bf16)(v * 0.125f);
          else if (MODE == 1)
            outB[((size_t)(b * NH + h) * SEQ + s) * DEP + d] = (__bf16)v;
          else
            outB[((size_t)(b * NH + h) * DEP + d) * SEQ + s] = (__bf16)v;
        }
      }
    }
  }
}

// one 16-query x 16-key logits tile (K = depth 64 -> 2 WMMAs)
static __device__ __forceinline__ v8f logits_tile(const __bf16* Kbase, int kt,
                                                  v16bf a0, v16bf a1, int g, int ln) {
  const __bf16* Krow = Kbase + (size_t)(kt * 16 + ln) * DEP;
  v16bf b0 = frag_split(Krow + 8 * g);
  v16bf b1 = frag_split(Krow + 32 + 8 * g);
  v8f c = {};
  c = wmma_bf16(a0, b0, c);
  c = wmma_bf16(a1, b1, c);
  return c;
}

// ---- fused attention: per 16-row query band, two-pass online softmax -----
// pass1: logits via WMMA (software-pipelined), online (max,sum), 1 exp/update
// pass2: recompute logits, write f32 softmax weights to d_out, stage bf16
//        weights through LDS, accumulate attn = W @ V via WMMA
__global__ __launch_bounds__(128) void k_attn(const __bf16* __restrict__ Qh,
                                              const __bf16* __restrict__ Kh,
                                              const __bf16* __restrict__ VhT,
                                              float* __restrict__ Wout,
                                              __bf16* __restrict__ concat) {
  __shared__ __bf16 smem[4 * 16 * 32];
  int wave = threadIdx.x >> 5, lane = threadIdx.x & 31;
  int g = lane >> 4, ln = lane & 15;
  int band = blockIdx.x * 4 + wave;        // 4096 bands = 32 (b,h) x 128 q-tiles
  int bh = band >> 7;
  int qt = band & 127;

  const __bf16* Qrow = Qh + ((size_t)bh * SEQ + qt * 16 + ln) * DEP;
  v16bf a0 = frag_split(Qrow + 8 * g);          // depth K = 0..31
  v16bf a1 = frag_split(Qrow + 32 + 8 * g);     // depth K = 32..63
  const __bf16* Kbase = Kh + (size_t)bh * SEQ * DEP;

  float m[8], sden[8];
#pragma unroll
  for (int i = 0; i < 8; ++i) { m[i] = -INFINITY; sden[i] = 0.f; }

  // ---------------- pass 1: online (max, sum) over all 2048 keys ----------
  // pipelined: produce tile kt+1 before consuming tile kt (hides WMMA->VALU)
  v8f cur = logits_tile(Kbase, 0, a0, a1, g, ln);
  for (int kt = 1; kt < SEQ / 16; ++kt) {
    v8f nxt = logits_tile(Kbase, kt, a0, a1, g, ln);
#pragma unroll
    for (int i = 0; i < 8; ++i) {
      float x = cur[i];
      float mn = fmaxf(m[i], x);
      float e = __expf(fminf(m[i], x) - mn);      // one exp: exp(-|x-m|)
      sden[i] = (x > m[i]) ? fmaf(sden[i], e, 1.f) : (sden[i] + e);
      m[i] = mn;
    }
    cur = nxt;
  }
#pragma unroll
  for (int i = 0; i < 8; ++i) {
    float x = cur[i];
    float mn = fmaxf(m[i], x);
    float e = __expf(fminf(m[i], x) - mn);
    sden[i] = (x > m[i]) ? fmaf(sden[i], e, 1.f) : (sden[i] + e);
    m[i] = mn;
  }
  // reduce (m,s) across the 16 lanes of each lane-group (row lives in 16 lanes)
#pragma unroll
  for (int off = 8; off >= 1; off >>= 1) {
#pragma unroll
    for (int i = 0; i < 8; ++i) {
      float mo = __shfl_xor(m[i], off, 32);
      float so = __shfl_xor(sden[i], off, 32);
      float mn = fmaxf(m[i], mo);
      float e = __expf(fminf(m[i], mo) - mn);     // one exp per merge
      sden[i] = (mo > m[i]) ? fmaf(sden[i], e, so) : fmaf(so, e, sden[i]);
      m[i] = mn;
    }
  }
  float inv[8];
#pragma unroll
  for (int i = 0; i < 8; ++i) inv[i] = 1.f / sden[i];

  // ---------------- pass 2: weights out + attn accumulation ---------------
  v8f o0 = {}, o1 = {}, o2 = {}, o3 = {};
  float* Wband = Wout + ((size_t)bh * SEQ + qt * 16) * SEQ;
  __bf16* myLds = smem + wave * (16 * 32);
  const __bf16* Vb = VhT + (size_t)bh * DEP * SEQ;

  for (int ks = 0; ks < SEQ / 32; ++ks) {
    int kbase = ks * 32;
    // produce both logit tiles before consuming either (distance for hazards)
    v8f c0 = logits_tile(Kbase, 2 * ks + 0, a0, a1, g, ln);
    v8f c1 = logits_tile(Kbase, 2 * ks + 1, a0, a1, g, ln);
#pragma unroll
    for (int t = 0; t < 2; ++t) {
      v8f cfr = t ? c1 : c0;
#pragma unroll
      for (int i = 0; i < 8; ++i) {
        int row = i + 8 * g;
        float w = __expf(cfr[i] - m[i]) * inv[i];
        Wband[(size_t)row * SEQ + kbase + t * 16 + ln] = w;  // f32 softmax out
        myLds[row * 32 + t * 16 + ln] = (__bf16)w;           // stage for A-frag
      }
    }
    asm volatile("s_wait_dscnt 0" ::: "memory");
    v16bf aw = frag_split(myLds + ln * 32 + 8 * g);          // 16x32 weight A-frag
#pragma unroll
    for (int t = 0; t < 4; ++t) {                            // depth tiles 0..63
      const __bf16* Vrow = Vb + (size_t)(t * 16 + ln) * SEQ + kbase + 8 * g;
      v16bf bv = frag_split(Vrow);
      if (t == 0) o0 = wmma_bf16(aw, bv, o0);
      else if (t == 1) o1 = wmma_bf16(aw, bv, o1);
      else if (t == 2) o2 = wmma_bf16(aw, bv, o2);
      else o3 = wmma_bf16(aw, bv, o3);
    }
  }

  // epilogue: attn -> concat bf16 [B,S,D] at column block h*64
  int b = bh >> 4, h = bh & 15;
  __bf16* Cc = concat + ((size_t)b * SEQ + qt * 16) * DM + h * DEP;
#pragma unroll
  for (int t = 0; t < 4; ++t) {
    v8f o = (t == 0) ? o0 : (t == 1) ? o1 : (t == 2) ? o2 : o3;
#pragma unroll
    for (int i = 0; i < 8; ++i)
      Cc[(size_t)(i + 8 * g) * DM + t * 16 + ln] = (__bf16)o[i];
  }
}

// ---------------------------------------------------------------------------
extern "C" void kernel_launch(void* const* d_in, const int* in_sizes, int n_in,
                              void* d_out, int out_size, void* d_ws, size_t ws_size,
                              hipStream_t stream) {
  const float* q   = (const float*)d_in[0];
  const float* k   = (const float*)d_in[1];
  const float* v   = (const float*)d_in[2];
  const float* w_q = (const float*)d_in[3];
  const float* b_q = (const float*)d_in[4];
  const float* w_k = (const float*)d_in[5];
  const float* b_k = (const float*)d_in[6];
  const float* w_v = (const float*)d_in[7];
  const float* b_v = (const float*)d_in[8];
  const float* w_o = (const float*)d_in[9];
  const float* b_o = (const float*)d_in[10];

  char* ws = (char*)d_ws;
  const size_t NE = (size_t)2 * SEQ * DM;           // 4,194,304 elements
  __bf16* qb   = (__bf16*)(ws);                     //  8 MiB each
  __bf16* kb   = (__bf16*)(ws + 8388608);
  __bf16* vb   = (__bf16*)(ws + 16777216);
  __bf16* wqT  = (__bf16*)(ws + 25165824);          //  2 MiB each
  __bf16* wkT  = (__bf16*)(ws + 27262976);
  __bf16* wvT  = (__bf16*)(ws + 29360128);
  __bf16* woT  = (__bf16*)(ws + 31457280);
  __bf16* Qh   = (__bf16*)(ws + 33554432);
  __bf16* Khp  = (__bf16*)(ws + 41943040);
  __bf16* VhT  = (__bf16*)(ws + 50331648);
  __bf16* conc = (__bf16*)(ws + 58720256);          // ends at 64 MiB

  float* outO = (float*)d_out;                      // [B,S,D]
  float* outW = outO + NE;                          // [B,H,S,S]

  // prep: bf16 copies of activations, transposed bf16 weights
  k_cvt_bf16<<<4096, 256, 0, stream>>>(q, qb);
  k_cvt_bf16<<<4096, 256, 0, stream>>>(k, kb);
  k_cvt_bf16<<<4096, 256, 0, stream>>>(v, vb);
  k_transpose_w<<<dim3(64, 64), dim3(16, 16), 0, stream>>>(w_q, wqT);
  k_transpose_w<<<dim3(64, 64), dim3(16, 16), 0, stream>>>(w_k, wkT);
  k_transpose_w<<<dim3(64, 64), dim3(16, 16), 0, stream>>>(w_v, wvT);
  k_transpose_w<<<dim3(64, 64), dim3(16, 16), 0, stream>>>(w_o, woT);

  // projections (scale 1/sqrt(64) folded into Q); 32x64 tile per wave
  k_gemm_proj<0><<<256, 256, 0, stream>>>(qb, wqT, b_q, Qh,  nullptr);
  k_gemm_proj<1><<<256, 256, 0, stream>>>(kb, wkT, b_k, Khp, nullptr);
  k_gemm_proj<2><<<256, 256, 0, stream>>>(vb, wvT, b_v, VhT, nullptr);

  // fused attention (softmax weights straight to d_out)
  k_attn<<<1024, 128, 0, stream>>>(Qh, Khp, VhT, outW, conc);

  // output projection
  k_gemm_proj<3><<<256, 256, 0, stream>>>(conc, woT, b_o, nullptr, outO);
}
